// eComformer_74277164417076
// MI455X (gfx1250) — matrix operand order repaired and checked by reading, not compiled, not run on hardware
//
// eComformer forward for MI455X (gfx1250), bf16 WMMA pipeline, v3.
//
// Roofline: ~65 GMAC total on <150MB live data (L2 = 192MB) -> compute bound.
// fp32 WMMA on CDNA5 is 16x16x4 only; bf16 WMMA is 16x16x32 (8x K density),
// so all GEMMs run as v_wmma_f32_16x16x32_bf16 with fp32 accumulate.
// Weights are stored transposed + K-padded in bf16 so the B fragment is a
// pair of 16B per-lane runs at k=8h and 16+8h (2x ds_load_b128) and staging is pure 16B
// vector copies; A fragments are two contiguous 16B runs per ISA 7.12.2.
// Each block computes a 32x64 tile: 4 waves x 2 WMMA per k-step, with LDS
// double-buffering + register prefetch (one s_barrier per k-step).
// Kernel is templated on (AMODE, EMODE, ACT) so each GEMM variant has a
// branch-free inner loop.
// The equi fc2 einsum is refactored: G[n,t,o] = sum_i W2[t,i*64+o]*s[n,i]
// (2.1 GMAC GEMM, 64MB bf16, L2-resident) replaces the naive 34 GMAC
// E x 256 x 4096 GEMM + 512MB intermediate.
//
// d_in layout (JAX pytree flatten = sorted dict keys):
//  0 x[2048,92]  1 edge_attr[32768,3]
//  2..27 conv0, 28..53 conv1, 54..79 conv2, each sorted:
//    +0..3 bn(g,b,m,v) +4..7 bn_att(g,b,m,v) +8,9 cat +10,11 e +12,13 k
//    +14,15 ku1 +16,17 ku2 +18,19 mu1 +20,21 mu2 +22,23 q +24,25 v
//  80,81 emb; 82..91 equi{fc1,fc2,nl,nl2,skip}; 92,93 fc; 94,95 fc_out;
//  96,97 rbf_lin; 98 edge_index[2,E] (src=row0,dst=row1); 99 batch[N]
#include <hip/hip_runtime.h>
#include <math.h>

typedef __bf16 bf16;
typedef __attribute__((ext_vector_type(4)))  __bf16 bf16x4;
typedef __attribute__((ext_vector_type(8)))  __bf16 bf16x8;
typedef __attribute__((ext_vector_type(16))) __bf16 bf16x16;
typedef __attribute__((ext_vector_type(8)))  float  f32x8;

#define ND 2048
#define NE 32768
#define NB 64
#define DD 256

__device__ __forceinline__ float softplusf(float x) {
  return (x > 20.f) ? x : log1pf(__expf(x));
}
__device__ __forceinline__ float sigmoidf_(float x) {
  return 1.f / (1.f + __expf(-x));
}

// ---------------------------------------------------------------- GEMM ----
// AMODE: 0=A bf16 [M,K] (lda elems; 4-elem groups in-bounds & 8B aligned)
//        1=A f32 [M,K] (*rowscale)   3=concat3 gather (0-255 P0[gidx0[m]],
//        256-511 P1[gidx1[m]], 512-767 P2[m])   4=analytic RBF from dvec[m]
// B: transposed K-padded bf16 weights, Bt[n][k], row stride ldb = Kpad.
// EMODE: 0=store (ACT: 0 none,1 silu,2 softplus,3 relu; wf32/wbf16 flags)
//        1=sigmoid(bn_att(q[dst]*kk/16)) gate + atomic scatter into scat[dst]
//        2=softplus(resid + bn(x)) -> Cf,Cb
//        3=softplus(x) + resid -> Cf,Cb
// Requires M % 32 == 0, N % 64 == 0 (true for all call sites).
struct GemmDesc {
  const bf16* A;  const float* Af; const float* rowscale;
  const bf16 *P0, *P1, *P2;
  const int *gidx0, *gidx1;
  const float* dvec;
  const bf16* B;
  const float* bias;
  float* Cf; bf16* Cb;
  const int* eidx;
  const float* qbuf; const float* kkbuf;
  const float *bng, *bnb, *bnm, *bnv;
  const float* resid;
  float* scat;
  int M, N, Kreal, lda, ldb, ldc;
  int amode, emode, act, wf32, wbf16;
};

template <int AMODE>
__device__ __forceinline__ void gatherA(const GemmDesc& g, int m0, int k0,
                                        int tid, bf16x4 av[2]) {
  const int c0 = (tid & 7) * 4;
  #pragma unroll
  for (int hrow = 0; hrow < 2; ++hrow) {
    const int r = (tid >> 3) + hrow * 16;
    const long m = m0 + r;
    const int gk = k0 + c0;
    bf16x4 v = {};
    if (AMODE == 0) {
      v = *(const bf16x4*)&g.A[m * g.lda + gk];
    } else if (AMODE == 1) {
      const float s = g.rowscale ? g.rowscale[m] : 1.f;
      #pragma unroll
      for (int j = 0; j < 4; ++j) {
        const float t = (gk + j < g.Kreal) ? g.Af[m * g.lda + gk + j] * s : 0.f;
        v[j] = (bf16)t;
      }
    } else if (AMODE == 3) {  // 4-elem group never straddles a 256-col segment
      const int seg = gk >> 8, cc = gk & 255;
      const bf16* p; long row;
      if (seg == 0)      { p = g.P0; row = g.gidx0[m]; }
      else if (seg == 1) { p = g.P1; row = g.gidx1[m]; }
      else               { p = g.P2; row = m; }
      v = *(const bf16x4*)&p[row * 256 + cc];
    } else if (AMODE == 4) {
      const float dd = g.dvec[m];
      #pragma unroll
      for (int j = 0; j < 4; ++j) {
        const float ck = -4.f + (4.f / 255.f) * (float)(gk + j);
        const float df = dd - ck;
        v[j] = (bf16)__expf(-63.75f * df * df);
      }
    }
    av[hrow] = v;
  }
}

__device__ __forceinline__ void loadB(const GemmDesc& g, int n0, int k0,
                                      int tid, bf16x8 bv[2]) {
  const int nl = tid >> 1;
  const int kc = (tid & 1) * 16;
  const bf16* sp = &g.B[(long)(n0 + nl) * g.ldb + k0 + kc];
  __builtin_prefetch(sp + 32, 0, 1);
  bv[0] = *(const bf16x8*)sp;
  bv[1] = *(const bf16x8*)(sp + 8);
}

template <int AMODE, int EMODE, int ACT>
__launch_bounds__(128)
__global__ void gemm32_wmma(GemmDesc g) {
  // 80B row stride keeps every 16B fragment run 16B-aligned.
  __shared__ bf16 As[2][32][40];
  __shared__ bf16 Bs[2][64][40];
  const int tid  = threadIdx.x;
  const int wv   = tid >> 5;      // wave id 0..3 (wave32)
  const int lane = tid & 31;
  const int m0 = blockIdx.x * 32;
  const int n0 = blockIdx.y * 64;
  const int hlf = lane >> 4;
  const int l15 = lane & 15;
  const int arow = tid >> 3, ac0 = (tid & 7) * 4;   // A staging coords
  const int bnl = tid >> 1, bkc = (tid & 1) * 16;   // B staging coords
  f32x8 acc0 = {}, acc1 = {};
  const int Kpad = (g.Kreal + 31) & ~31;

  bf16x4 av[2]; bf16x8 bv[2];
  gatherA<AMODE>(g, m0, 0, tid, av);
  loadB(g, n0, 0, tid, bv);
  *(bf16x4*)&As[0][arow][ac0]      = av[0];
  *(bf16x4*)&As[0][arow + 16][ac0] = av[1];
  *(bf16x8*)&Bs[0][bnl][bkc]       = bv[0];
  *(bf16x8*)&Bs[0][bnl][bkc + 8]   = bv[1];
  __syncthreads();

  for (int k0 = 0; k0 < Kpad; k0 += 32) {
    const int cur = (k0 >> 5) & 1;
    const bool more = (k0 + 32) < Kpad;
    if (more) {  // register prefetch of next tile (overlaps WMMA below)
      gatherA<AMODE>(g, m0, k0 + 32, tid, av);
      loadB(g, n0, k0 + 32, tid, bv);
    }
    // ---- fragments (ISA 7.12.2), contiguous 16B runs -> ds_load_b128
    const int col = wv * 16 + l15;
    bf16x8 b0 = *(const bf16x8*)&Bs[cur][col][hlf * 8];
    bf16x8 b1 = *(const bf16x8*)&Bs[cur][col][hlf * 8 + 16];
    bf16x16 bfr = __builtin_shufflevector(
        b0, b1, 0, 1, 2, 3, 4, 5, 6, 7, 8, 9, 10, 11, 12, 13, 14, 15);
    bf16x8 a00 = *(const bf16x8*)&As[cur][l15][hlf * 8];
    bf16x8 a01 = *(const bf16x8*)&As[cur][l15][hlf * 8 + 16];
    bf16x16 af0 = __builtin_shufflevector(
        a00, a01, 0, 1, 2, 3, 4, 5, 6, 7, 8, 9, 10, 11, 12, 13, 14, 15);
    bf16x8 a10 = *(const bf16x8*)&As[cur][16 + l15][hlf * 8];
    bf16x8 a11 = *(const bf16x8*)&As[cur][16 + l15][hlf * 8 + 16];
    bf16x16 af1 = __builtin_shufflevector(
        a10, a11, 0, 1, 2, 3, 4, 5, 6, 7, 8, 9, 10, 11, 12, 13, 14, 15);
    acc0 = __builtin_amdgcn_wmma_f32_16x16x32_bf16(
        false, af0, false, bfr, (short)0, acc0, false, false);
    acc1 = __builtin_amdgcn_wmma_f32_16x16x32_bf16(
        false, af1, false, bfr, (short)0, acc1, false, false);
    if (more) {  // commit prefetched tile to the alternate LDS buffer
      const int nxt = cur ^ 1;
      *(bf16x4*)&As[nxt][arow][ac0]      = av[0];
      *(bf16x4*)&As[nxt][arow + 16][ac0] = av[1];
      *(bf16x8*)&Bs[nxt][bnl][bkc]       = bv[0];
      *(bf16x8*)&Bs[nxt][bnl][bkc + 8]   = bv[1];
    }
    __syncthreads();
  }
  // ---- epilogue; C/D: VGPR r -> M = r + 8*hlf, N = l15
  #pragma unroll
  for (int sub = 0; sub < 2; ++sub) {
    const f32x8 acc = sub ? acc1 : acc0;
    #pragma unroll
    for (int r = 0; r < 8; ++r) {
      const int m = m0 + sub * 16 + r + hlf * 8;
      const int n = n0 + wv * 16 + l15;
      float x = acc[r];
      if (g.bias) x += g.bias[n];
      if (EMODE == 0) {
        if (ACT == 1)      x = x * sigmoidf_(x);
        else if (ACT == 2) x = softplusf(x);
        else if (ACT == 3) x = fmaxf(x, 0.f);
        if (g.wf32)  g.Cf[(long)m * g.ldc + n] = x;
        if (g.wbf16) g.Cb[(long)m * g.ldc + n] = (bf16)x;
      } else if (EMODE == 1) {  // gated message scatter (aggr by dst)
        const int dn = g.eidx[m];
        const float a = g.qbuf[(long)dn * 256 + n] *
                        g.kkbuf[(long)m * 256 + n] * 0.0625f;  // 1/sqrt(D)
        const float z = (a - g.bnm[n]) * rsqrtf(g.bnv[n] + 1e-5f) *
                        g.bng[n] + g.bnb[n];
        atomicAdd(&g.scat[(long)dn * 256 + n], x * sigmoidf_(z));
      } else if (EMODE == 2) {  // softplus(h + bn(x))
        float z = (x - g.bnm[n]) * rsqrtf(g.bnv[n] + 1e-5f) * g.bng[n] +
                  g.bnb[n];
        z = softplusf(z + g.resid[(long)m * g.ldc + n]);
        g.Cf[(long)m * g.ldc + n] = z;
        g.Cb[(long)m * g.ldc + n] = (bf16)z;
      } else {  // EMODE 3: softplus(x) + skip
        const float z = softplusf(x) + g.resid[(long)m * g.ldc + n];
        g.Cf[(long)m * g.ldc + n] = z;
        g.Cb[(long)m * g.ldc + n] = (bf16)z;
      }
    }
  }
}

// ---------------------------------------------------------- small kernels
// transpose + K-pad: d[n*Kpad + k] = (k<K) ? s[k*N + n] : 0
__global__ void k_cvt_t(const float* s, bf16* d, int K, int Kpad, int N) {
  long i = (long)blockIdx.x * blockDim.x + threadIdx.x;
  if (i >= (long)N * Kpad) return;
  const int n = (int)(i / Kpad), k = (int)(i % Kpad);
  d[i] = (k < K) ? (bf16)s[(long)k * N + n] : (bf16)0.f;
}
__global__ void k_zero_f32(float* p, long n) {
  long i = (long)blockIdx.x * blockDim.x + threadIdx.x;
  if (i < n) p[i] = 0.f;
}
__global__ void k_edge_d(const float* ea, float* dv, int E) {
  int e = blockIdx.x * blockDim.x + threadIdx.x;
  if (e < E) {
    float x = ea[e * 3], y = ea[e * 3 + 1], z = ea[e * 3 + 2];
    dv[e] = -0.75f * rsqrtf(x * x + y * y + z * z);
  }
}
// W2rT[(t*64+o)*64 + i] = Wfc2[t*5120 + i*64 + o]  (transposed G weight)
__global__ void k_permute_fc2T(const float* w, bf16* out) {
  long i = (long)blockIdx.x * blockDim.x + threadIdx.x;
  if (i >= (long)16384 * 64) return;
  const int col = (int)(i >> 6);       // t*64+o
  const int ii  = (int)(i & 63);       // input index i
  const int t = col >> 6, o = col & 63;
  out[i] = (bf16)w[(long)t * 5120 + ii * 64 + o];
}
__global__ void k_cnt_src(const int* src, float* cnt, int E) {
  int e = blockIdx.x * blockDim.x + threadIdx.x;
  if (e < E) atomicAdd(&cnt[src[e]], 1.f);
}
__global__ void k_recip(const float* cnt, float* rs, int n) {
  int i = blockIdx.x * blockDim.x + threadIdx.x;
  if (i < n) rs[i] = 1.f / fmaxf(cnt[i], 1.f);
}
// out0[e,o] = (sum_t ww[e,t]*G[dst[e],t,o] + bs[dst[e],o]) / 8 ; scatter->src
__launch_bounds__(256)
__global__ void k_equi_edge(const bf16* __restrict__ ww,
                            const bf16* __restrict__ G,
                            const float* __restrict__ bs,
                            const int* __restrict__ src,
                            const int* __restrict__ dst,
                            float* __restrict__ aggE) {
  __shared__ float sw[256];
  __shared__ float red[16][64];
  const int e = blockIdx.x, t = threadIdx.x;
  sw[t] = (float)ww[(long)e * 256 + t];
  __syncthreads();
  const int dn = dst[e], sn = src[e];
  const int part = t >> 4;          // k-chunk 0..15
  const int og   = (t & 15) * 4;    // 4 outputs per thread
  const bf16* Gp = G + (long)dn * 16384;
  float a0 = 0.f, a1 = 0.f, a2 = 0.f, a3 = 0.f;
  #pragma unroll
  for (int q = 0; q < 16; ++q) {
    const int tt = part * 16 + q;
    const float sv = sw[tt];
    const bf16x4 gv = *(const bf16x4*)&Gp[tt * 64 + og];
    a0 += sv * (float)gv[0]; a1 += sv * (float)gv[1];
    a2 += sv * (float)gv[2]; a3 += sv * (float)gv[3];
  }
  red[part][og] = a0; red[part][og + 1] = a1;
  red[part][og + 2] = a2; red[part][og + 3] = a3;
  __syncthreads();
  if (t < 64) {
    float v = 0.f;
    #pragma unroll
    for (int p = 0; p < 16; ++p) v += red[p][t];
    v = (v + bs[(long)dn * 64 + t]) * 0.125f;  // / sqrt(NS)
    atomicAdd(&aggE[(long)sn * 64 + t], v);
  }
}
__global__ void k_pool_scatter(const float* h, const int* batch,
                               float* gsum, float* bcnt) {
  const int n = blockIdx.x, t = threadIdx.x;
  const int b = batch[n];
  atomicAdd(&gsum[(long)b * 256 + t], h[(long)n * 256 + t]);
  if (t == 0) atomicAdd(&bcnt[b], 1.f);
}
__global__ void k_pool_finish(const float* gsum, const float* bcnt, bf16* gp) {
  const int b = blockIdx.x, t = threadIdx.x;
  gp[(long)b * 256 + t] = (bf16)(gsum[(long)b * 256 + t] / fmaxf(bcnt[b], 1.f));
}
__global__ void k_fcout(const float* gf, const float* w, const float* b0,
                        float* out) {
  __shared__ float red[256];
  const int b = blockIdx.x, t = threadIdx.x;
  red[t] = gf[(long)b * 256 + t] * w[t];
  __syncthreads();
  for (int s = 128; s > 0; s >>= 1) {
    if (t < s) red[t] += red[t + s];
    __syncthreads();
  }
  if (t == 0) out[b] = red[0] + b0[0];
}

// ------------------------------------------------------------------ host
extern "C" void kernel_launch(void* const* d_in, const int* in_sizes, int n_in,
                              void* d_out, int out_size, void* d_ws,
                              size_t ws_size, hipStream_t stream) {
  (void)in_sizes; (void)n_in; (void)out_size;
  if (ws_size < (size_t)128 << 20) return;  // needs ~120MB scratch

  const float* x  = (const float*)d_in[0];
  const float* ea = (const float*)d_in[1];
  const int* ei   = (const int*)d_in[98];
  const int* src  = ei;
  const int* dst  = ei + NE;
  const int* batch = (const int*)d_in[99];
  auto F = [&](int i) { return (const float*)d_in[i]; };
  const int PC[3] = {2, 28, 54};

  // bump allocator over d_ws
  char* wp = (char*)d_ws;
  auto alloc = [&](size_t bytes) -> char* {
    char* r = wp; wp += (bytes + 255) & ~(size_t)255; return r;
  };
  float* h    = (float*)alloc((size_t)ND * DD * 4);
  bf16* hb    = (bf16*)alloc((size_t)ND * DD * 2);
  bf16* efb   = (bf16*)alloc((size_t)NE * DD * 2);
  bf16* kb_   = (bf16*)alloc((size_t)ND * DD * 2);
  bf16* vb_   = (bf16*)alloc((size_t)ND * DD * 2);
  float* qf   = (float*)alloc((size_t)ND * DD * 4);
  // overlay region: conv uses kk(f32)+tbuf(bf16)+e_bf(bf16); equi uses G(bf16)
  char* big   = alloc((size_t)ND * 16384 * 2);  // 64MB
  float* kk   = (float*)big;
  bf16* tb    = (bf16*)(big + (size_t)NE * DD * 4);
  bf16* eb    = (bf16*)(big + (size_t)NE * DD * 4 + (size_t)NE * DD * 2);
  bf16* G     = (bf16*)big;
  bf16* wwb   = (bf16*)alloc((size_t)NE * DD * 2);
  float* agg  = (float*)alloc((size_t)ND * DD * 4);
  bf16* sb    = (bf16*)alloc((size_t)ND * 64 * 2);
  float* bsb  = (float*)alloc((size_t)ND * 64 * 4);
  float* aggE = (float*)alloc((size_t)ND * 64 * 4);
  float* cnt  = (float*)alloc((size_t)ND * 4);
  float* rs   = (float*)alloc((size_t)ND * 4);
  float* skp  = (float*)alloc((size_t)ND * DD * 4);
  float* dv   = (float*)alloc((size_t)NE * 4);
  float* gsum = (float*)alloc((size_t)NB * DD * 4);
  float* bcnt = (float*)alloc((size_t)NB * 4);
  bf16* gpb   = (bf16*)alloc((size_t)NB * DD * 2);
  float* gff  = (float*)alloc((size_t)NB * DD * 4);

  auto cvtT = [&](const float* s, bf16* d, int K, int Kpad, int N) {
    long n = (long)N * Kpad;
    k_cvt_t<<<dim3((unsigned)((n + 255) / 256)), dim3(256), 0, stream>>>(
        s, d, K, Kpad, N);
  };
  auto zero = [&](float* p, long n) {
    k_zero_f32<<<dim3((unsigned)((n + 255) / 256)), dim3(256), 0, stream>>>(p, n);
  };
  auto run = [&](const GemmDesc& d) {
    dim3 grid((unsigned)(d.M / 32), (unsigned)(d.N / 64));
#define GCASE(A_, E_, AC_)                                            \
  if (d.amode == A_ && d.emode == E_ && d.act == AC_) {               \
    gemm32_wmma<A_, E_, AC_><<<grid, dim3(128), 0, stream>>>(d);      \
    return;                                                           \
  }
    GCASE(0, 0, 0) GCASE(0, 0, 1) GCASE(0, 0, 3) GCASE(0, 1, 0)
    GCASE(1, 0, 0) GCASE(1, 2, 0) GCASE(1, 3, 0) GCASE(3, 0, 1)
    GCASE(4, 0, 2)
#undef GCASE
  };

  // ---- weights -> bf16, transposed [N][Kpad]
  bf16* Wemb = (bf16*)alloc((size_t)256 * 96 * 2);
  cvtT(F(80), Wemb, 92, 96, 256);
  bf16* Wrbf = (bf16*)alloc((size_t)256 * 256 * 2);
  cvtT(F(96), Wrbf, 256, 256, 256);
  // per conv: q,k,v,e,ku1,ku2,mu1,mu2,cat (param offsets vs leaf list)
  const int woff[9] = {22, 12, 24, 10, 14, 16, 18, 20, 8};
  const int wK[9]   = {256, 256, 256, 256, 768, 256, 768, 256, 256};
  bf16* Wc[3][9];
  for (int l = 0; l < 3; ++l)
    for (int j = 0; j < 9; ++j) {
      Wc[l][j] = (bf16*)alloc((size_t)256 * wK[j] * 2);
      cvtT(F(PC[l] + woff[j]), Wc[l][j], wK[j], wK[j], 256);
    }
  bf16* Wnl  = (bf16*)alloc((size_t)64 * 256 * 2);
  cvtT(F(86), Wnl, 256, 256, 64);
  bf16* Wfc1 = (bf16*)alloc((size_t)256 * 256 * 2);
  cvtT(F(82), Wfc1, 256, 256, 256);
  bf16* Wskp = (bf16*)alloc((size_t)256 * 256 * 2);
  cvtT(F(90), Wskp, 256, 256, 256);
  bf16* Wnl2 = (bf16*)alloc((size_t)256 * 64 * 2);
  cvtT(F(88), Wnl2, 64, 64, 256);
  bf16* W2rT = (bf16*)alloc((size_t)16384 * 64 * 2);
  k_permute_fc2T<<<dim3(4096), dim3(256), 0, stream>>>(F(84), W2rT);
  bf16* B2rT = (bf16*)alloc((size_t)64 * 64 * 2);
  cvtT(F(85), B2rT, 64, 64, 64);   // B2r[i][o]=b2[i*64+o]; transposed
  bf16* Wfc  = (bf16*)alloc((size_t)256 * 256 * 2);
  cvtT(F(92), Wfc, 256, 256, 256);

  // ---- stage 0: d, embedding, edge features
  k_edge_d<<<dim3(NE / 256), dim3(256), 0, stream>>>(ea, dv, NE);
  { GemmDesc d{}; d.amode = 1; d.Af = x; d.lda = 92; d.Kreal = 92;
    d.B = Wemb; d.ldb = 96; d.bias = F(81); d.M = ND; d.N = 256;
    d.wf32 = 1; d.wbf16 = 1; d.Cf = h; d.Cb = hb; d.ldc = 256; run(d); }
  { GemmDesc d{}; d.amode = 4; d.dvec = dv; d.Kreal = 256;
    d.B = Wrbf; d.ldb = 256; d.bias = F(97); d.M = NE; d.N = 256;
    d.act = 2; d.wbf16 = 1; d.Cb = efb; d.ldc = 256; run(d); }

  // ---- conv layers (equi after conv0)
  for (int l = 0; l < 3; ++l) {
    const int p = PC[l];
    { GemmDesc d{}; d.A = hb; d.lda = 256; d.Kreal = 256; d.B = Wc[l][0];
      d.ldb = 256; d.bias = F(p + 23); d.M = ND; d.N = 256; d.wf32 = 1;
      d.Cf = qf; d.ldc = 256; run(d); }                           // q (f32)
    { GemmDesc d{}; d.A = hb; d.lda = 256; d.Kreal = 256; d.B = Wc[l][1];
      d.ldb = 256; d.bias = F(p + 13); d.M = ND; d.N = 256; d.wbf16 = 1;
      d.Cb = kb_; d.ldc = 256; run(d); }                          // k
    { GemmDesc d{}; d.A = hb; d.lda = 256; d.Kreal = 256; d.B = Wc[l][2];
      d.ldb = 256; d.bias = F(p + 25); d.M = ND; d.N = 256; d.wbf16 = 1;
      d.Cb = vb_; d.ldc = 256; run(d); }                          // v
    { GemmDesc d{}; d.A = efb; d.lda = 256; d.Kreal = 256; d.B = Wc[l][3];
      d.ldb = 256; d.bias = F(p + 11); d.M = NE; d.N = 256; d.wbf16 = 1;
      d.Cb = eb; d.ldc = 256; run(d); }                           // e
    { GemmDesc d{}; d.amode = 3; d.P0 = kb_; d.P1 = kb_; d.P2 = eb;
      d.gidx0 = dst; d.gidx1 = src; d.Kreal = 768; d.B = Wc[l][4];
      d.ldb = 768; d.bias = F(p + 15); d.M = NE; d.N = 256; d.act = 1;
      d.wbf16 = 1; d.Cb = tb; d.ldc = 256; run(d); }              // ku1+silu
    { GemmDesc d{}; d.A = tb; d.lda = 256; d.Kreal = 256; d.B = Wc[l][5];
      d.ldb = 256; d.bias = F(p + 17); d.M = NE; d.N = 256; d.wf32 = 1;
      d.Cf = kk; d.ldc = 256; run(d); }                           // ku2 -> kk
    { GemmDesc d{}; d.amode = 3; d.P0 = vb_; d.P1 = vb_; d.P2 = eb;
      d.gidx0 = dst; d.gidx1 = src; d.Kreal = 768; d.B = Wc[l][6];
      d.ldb = 768; d.bias = F(p + 19); d.M = NE; d.N = 256; d.act = 1;
      d.wbf16 = 1; d.Cb = tb; d.ldc = 256; run(d); }              // mu1+silu
    zero(agg, (long)ND * 256);
    { GemmDesc d{}; d.A = tb; d.lda = 256; d.Kreal = 256; d.B = Wc[l][7];
      d.ldb = 256; d.bias = F(p + 21); d.M = NE; d.N = 256; d.emode = 1;
      d.eidx = dst; d.qbuf = qf; d.kkbuf = kk; d.bng = F(p + 4);
      d.bnb = F(p + 5); d.bnm = F(p + 6); d.bnv = F(p + 7);
      d.scat = agg; d.ldc = 256; run(d); }              // mu2 + gate + scatter
    { GemmDesc d{}; d.amode = 1; d.Af = agg; d.lda = 256; d.Kreal = 256;
      d.B = Wc[l][8]; d.ldb = 256; d.bias = F(p + 9); d.M = ND; d.N = 256;
      d.emode = 2; d.bng = F(p + 0); d.bnb = F(p + 1); d.bnm = F(p + 2);
      d.bnv = F(p + 3); d.resid = h; d.Cf = h; d.Cb = hb; d.ldc = 256;
      run(d); }                                          // cat+bn+res+softplus

    if (l == 0) {  // ---- equi block
      { GemmDesc d{}; d.A = hb; d.lda = 256; d.Kreal = 256; d.B = Wnl;
        d.ldb = 256; d.bias = F(87); d.M = ND; d.N = 64; d.wbf16 = 1;
        d.Cb = sb; d.ldc = 64; run(d); }                           // s
      { GemmDesc d{}; d.A = efb; d.lda = 256; d.Kreal = 256; d.B = Wfc1;
        d.ldb = 256; d.bias = F(83); d.M = NE; d.N = 256; d.act = 3;
        d.wbf16 = 1; d.Cb = wwb; d.ldc = 256; run(d); }            // relu(fc1)
      { GemmDesc d{}; d.A = sb; d.lda = 64; d.Kreal = 64; d.B = W2rT;
        d.ldb = 64; d.M = ND; d.N = 16384; d.wbf16 = 1; d.Cb = G;
        d.ldc = 16384; run(d); }                                   // G tensor
      { GemmDesc d{}; d.A = sb; d.lda = 64; d.Kreal = 64; d.B = B2rT;
        d.ldb = 64; d.M = ND; d.N = 64; d.wf32 = 1; d.Cf = bsb;
        d.ldc = 64; run(d); }                                      // bias term
      zero(aggE, (long)ND * 64); zero(cnt, ND);
      k_cnt_src<<<dim3(NE / 256), dim3(256), 0, stream>>>(src, cnt, NE);
      k_recip<<<dim3(ND / 256), dim3(256), 0, stream>>>(cnt, rs, ND);
      k_equi_edge<<<dim3(NE), dim3(256), 0, stream>>>(wwb, G, bsb, src, dst,
                                                      aggE);
      { GemmDesc d{}; d.A = hb; d.lda = 256; d.Kreal = 256; d.B = Wskp;
        d.ldb = 256; d.bias = F(91); d.M = ND; d.N = 256; d.wf32 = 1;
        d.Cf = skp; d.ldc = 256; run(d); }                         // skip
      { GemmDesc d{}; d.amode = 1; d.Af = aggE; d.lda = 64; d.Kreal = 64;
        d.rowscale = rs; d.B = Wnl2; d.ldb = 64; d.bias = F(89); d.M = ND;
        d.N = 256; d.emode = 3; d.resid = skp; d.Cf = h; d.Cb = hb;
        d.ldc = 256; run(d); }                        // softplus(nl2) + skip
    }
  }

  // ---- readout
  zero(gsum, (long)NB * 256); zero(bcnt, NB);
  k_pool_scatter<<<dim3(ND), dim3(256), 0, stream>>>(h, batch, gsum, bcnt);
  k_pool_finish<<<dim3(NB), dim3(256), 0, stream>>>(gsum, bcnt, gpb);
  { GemmDesc d{}; d.A = gpb; d.lda = 256; d.Kreal = 256; d.B = Wfc;
    d.ldb = 256; d.bias = F(93); d.M = NB; d.N = 256; d.act = 1; d.wf32 = 1;
    d.Cf = gff; d.ldc = 256; run(d); }
  k_fcout<<<dim3(NB), dim3(256), 0, stream>>>(gff, F(94), F(95),
                                              (float*)d_out);
}
